// MambaSS2D_28132035788838
// MI455X (gfx1250) — matrix operand order repaired and checked by reading, not compiled, not run on hardware
//
#include <hip/hip_runtime.h>
#include <hip/hip_bf16.h>

#define DEV __device__ __forceinline__

typedef __bf16 v16bf __attribute__((ext_vector_type(16)));
typedef float  v8f   __attribute__((ext_vector_type(8)));

// ---- problem constants ----
constexpr int D_MODEL = 96;
constexpr int D_INNER = 192;
constexpr int D_STATE = 16;
constexpr int DT_RANK = 6;
constexpr int ND      = DT_RANK + 2 * D_STATE;  // 38
constexpr int NDP     = 48;                     // padded to 3 n-tiles
constexpr int BATCH   = 2;
constexpr int Hs      = 128;
constexpr int Wsz     = 128;
constexpr int L       = Hs * Wsz;               // 16384
constexpr int CHUNK   = 64;
constexpr int NCHUNK  = L / CHUNK;              // 256
constexpr int ROWS    = CHUNK + 16;             // 16-row lookback for conv taps

// ---- workspace layout (bytes) ----
constexpr size_t WS_H_OFF  = 0;
constexpr size_t WS_H_SZ   = (size_t)4 * BATCH * NCHUNK * D_INNER * D_STATE * sizeof(float);
constexpr size_t WS_DT_OFF = WS_H_OFF + WS_H_SZ;
constexpr size_t WS_DT_SZ  = (size_t)4 * BATCH * NCHUNK * D_INNER * sizeof(float);
constexpr size_t WS_Y_OFF  = WS_DT_OFF + WS_DT_SZ;   // bf16 (BATCH, L, 384)

// ---- dynamic LDS layout (bytes) ----
constexpr int SM_DT   = 0;                             // f32  [64][192]
constexpr int SM_XDBL = SM_DT   + CHUNK * D_INNER * 4; // f32  [64][48]
constexpr int SM_XBF  = SM_XDBL + CHUNK * NDP    * 4;  // bf16 [80][96]
constexpr int SM_XC   = SM_XBF  + ROWS  * D_MODEL * 2; // bf16 [80][192]
constexpr int SM_U    = SM_XC   + ROWS  * D_INNER * 2; // bf16 [64][192] (reused as y)
constexpr int SM_Z    = SM_U    + CHUNK * D_INNER * 2; // bf16 [64][192] (pass C only)
constexpr int SMEM_A  = SM_Z;
constexpr int SMEM_C  = SM_Z + CHUNK * D_INNER * 2;

DEV float siluf(float x)     { return x / (1.0f + __expf(-x)); }
DEV float softplusf(float x) { return (x > 20.0f) ? x : log1pf(__expf(x)); }

// scan-order position l -> flat spatial index p (same map for input & output)
DEV int permP(int k, int l) {
  int ll = (k & 1) ? (L - 1 - l) : l;
  if (k & 2) ll = ((ll & (Wsz - 1)) << 7) | (ll >> 7);  // 128x128 transpose
  return ll;
}

// 16-bit A-matrix 16x32 fragment from row-major bf16 LDS buffer.
// lane: row M = lane&15; K = kbase + (lane>>4)*8 + {0..7, 16..23}
DEV v16bf load_afrag_lds(const __bf16* A, int ld, int mbase, int kbase, int lane) {
  const __bf16* p = A + (mbase + (lane & 15)) * ld + kbase + ((lane >> 4) << 3);
  v16bf r;
#pragma unroll
  for (int e = 0; e < 8; ++e) { r[e] = p[e]; r[e + 8] = p[e + 16]; }
  return r;
}

// B-matrix 32x16 fragment from row-major f32 weight W[n][kk] (K contiguous).
// lane: col N = lane&15; K = kbase + (lane>>4)*8 + {0..7, 16..23}
DEV v16bf load_bfrag_w(const float* W, int ldK, int nrows, int nbase, int kbase, int lane) {
  int n = nbase + (lane & 15);
  v16bf r;
  if (n < nrows) {
    const float* p = W + (size_t)n * ldK + kbase + ((lane >> 4) << 3);
#pragma unroll
    for (int e = 0; e < 8; ++e) { r[e] = (__bf16)p[e]; r[e + 8] = (__bf16)p[e + 16]; }
  } else {
#pragma unroll
    for (int e = 0; e < 16; ++e) r[e] = (__bf16)0.0f;
  }
  return r;
}

DEV v8f wmma_bf16(v16bf a, v16bf b, v8f c) {
  return __builtin_amdgcn_wmma_f32_16x16x32_bf16(false, a, false, b, (short)0, c, false, false);
}

// ---- shared front-end: stage x, in-proj, conv+silu, x-proj, dt ----
DEV void chunk_frontend(int k, int b, int c, bool need_z,
                        const float* x, const float* in_w, const float* conv_w,
                        const float* conv_b, const float* xproj_w,
                        const float* dt_w, const float* dt_b,
                        __bf16* s_xbf, __bf16* s_xc, __bf16* s_u,
                        float* s_dt, float* s_xdbl, __bf16* s_z) {
  const int tid = threadIdx.x;
  const int lane = tid & 31;
  const int wv = tid >> 5;

  // 1) stage x rows in scan order (bf16); 16-row lookback, zero-padded at l<0
  for (int idx = tid; idx < ROWS * D_MODEL; idx += 256) {
    int rr = idx / D_MODEL, ch = idx % D_MODEL;
    int l = c * CHUNK - 16 + rr;
    float v = 0.0f;
    if (l >= 0) v = x[((size_t)b * D_MODEL + ch) * L + permP(k, l)];
    s_xbf[rr * D_MODEL + ch] = (__bf16)v;
  }
  __syncthreads();

  // 2) in-proj GEMM: (80x96) @ (96 -> 192 [+192 z]) via WMMA bf16
  const float* Wk = in_w + (size_t)k * 2 * D_INNER * D_MODEL;
  const int ntiles = need_z ? 24 : 12;
  for (int j = wv; j < 5 * ntiles; j += 8) {
    int mt = j / ntiles, nt = j % ntiles;
    v8f acc = {};
#pragma unroll
    for (int kt = 0; kt < 3; ++kt) {
      v16bf a  = load_afrag_lds(s_xbf, D_MODEL, mt * 16, kt * 32, lane);
      v16bf bb = load_bfrag_w(Wk, D_MODEL, 2 * D_INNER, nt * 16, kt * 32, lane);
      acc = wmma_bf16(a, bb, acc);
    }
    int M0 = (lane >> 4) << 3, N = lane & 15;
#pragma unroll
    for (int r = 0; r < 8; ++r) {
      int row = mt * 16 + M0 + r;
      float v = acc[r];
      if (nt < 12) s_xc[row * D_INNER + nt * 16 + N] = (__bf16)v;
      else if (row >= 16) s_z[(row - 16) * D_INNER + (nt - 12) * 16 + N] = (__bf16)siluf(v);
    }
  }
  __syncthreads();

  // 3) depthwise causal conv(4) + bias + silu -> u
  for (int idx = tid; idx < CHUNK * D_INNER; idx += 256) {
    int t = idx / D_INNER, d = idx % D_INNER;
    int rr = t + 16;
    float acc = conv_b[k * D_INNER + d];
#pragma unroll
    for (int j = 0; j < 4; ++j)
      acc += conv_w[((size_t)k * D_INNER + d) * 4 + j] * (float)s_xc[(rr - 3 + j) * D_INNER + d];
    s_u[t * D_INNER + d] = (__bf16)siluf(acc);
  }
  __syncthreads();

  // 4) x-proj GEMM: (64x192) @ (192 -> 38, padded 48)
  const float* Wx = xproj_w + (size_t)k * ND * D_INNER;
  for (int j = wv; j < 12; j += 8) {
    int mt = j / 3, nt = j % 3;
    v8f acc = {};
#pragma unroll
    for (int kt = 0; kt < 6; ++kt) {
      v16bf a  = load_afrag_lds(s_u, D_INNER, mt * 16, kt * 32, lane);
      v16bf bb = load_bfrag_w(Wx, D_INNER, ND, nt * 16, kt * 32, lane);
      acc = wmma_bf16(a, bb, acc);
    }
    int M0 = (lane >> 4) << 3, N = lane & 15;
#pragma unroll
    for (int r = 0; r < 8; ++r)
      s_xdbl[(mt * 16 + M0 + r) * NDP + nt * 16 + N] = acc[r];
  }
  __syncthreads();

  // 5) dt = softplus(dt_raw @ dt_w.T + dt_b)  (K=6, VALU)
  for (int idx = tid; idx < CHUNK * D_INNER; idx += 256) {
    int t = idx / D_INNER, d = idx % D_INNER;
    float acc = dt_b[k * D_INNER + d];
#pragma unroll
    for (int r = 0; r < DT_RANK; ++r)
      acc += s_xdbl[t * NDP + r] * dt_w[((size_t)k * D_INNER + d) * DT_RANK + r];
    s_dt[t * D_INNER + d] = softplusf(acc);
  }
  __syncthreads();
}

// ---- pass A: per-chunk local state (h with h_in=0) and sum(dt) ----
__global__ void k_chunk_state(const float* x, const float* in_w, const float* conv_w,
                              const float* conv_b, const float* xproj_w,
                              const float* dt_w, const float* dt_b, const float* A_log,
                              float* ws_h, float* ws_dtsum) {
  extern __shared__ char smem[];
  float*  s_dt   = (float*)(smem + SM_DT);
  float*  s_xdbl = (float*)(smem + SM_XDBL);
  __bf16* s_xbf  = (__bf16*)(smem + SM_XBF);
  __bf16* s_xc   = (__bf16*)(smem + SM_XC);
  __bf16* s_u    = (__bf16*)(smem + SM_U);

  int c = blockIdx.x, kb = blockIdx.y;
  int k = kb >> 1, b = kb & 1;
  chunk_frontend(k, b, c, false, x, in_w, conv_w, conv_b, xproj_w, dt_w, dt_b,
                 s_xbf, s_xc, s_u, s_dt, s_xdbl, nullptr);

  int d = threadIdx.x;
  if (d < D_INNER) {
    float As[D_STATE], h[D_STATE];
#pragma unroll
    for (int s = 0; s < D_STATE; ++s) {
      As[s] = -__expf(A_log[((size_t)k * D_INNER + d) * D_STATE + s]);
      h[s] = 0.0f;
    }
    float dtsum = 0.0f;
    for (int t = 0; t < CHUNK; ++t) {
      float dtv = s_dt[t * D_INNER + d];
      float uv  = (float)s_u[t * D_INNER + d];
      dtsum += dtv;
      float du = dtv * uv;
#pragma unroll
      for (int s = 0; s < D_STATE; ++s) {
        float Bs = s_xdbl[t * NDP + DT_RANK + s];
        h[s] = __expf(dtv * As[s]) * h[s] + du * Bs;
      }
    }
    size_t base = (((size_t)kb * NCHUNK) + c) * D_INNER + d;
#pragma unroll
    for (int s = 0; s < D_STATE; ++s) ws_h[base * D_STATE + s] = h[s];
    ws_dtsum[base] = dtsum;
  }
}

// ---- pass B: sequential inter-chunk combine; ws_h becomes carry-in state ----
__global__ void k_combine(const float* A_log, float* ws_h, const float* ws_dtsum) {
  int gid = blockIdx.x * blockDim.x + threadIdx.x;
  if (gid >= 8 * D_INNER * D_STATE) return;
  int kb = gid / (D_INNER * D_STATE);
  int ds = gid % (D_INNER * D_STATE);
  int d = ds >> 4, s = ds & 15;
  int k = kb >> 1;
  float As = -__expf(A_log[((size_t)k * D_INNER + d) * D_STATE + s]);
  float h = 0.0f;
  for (int c = 0; c < NCHUNK; ++c) {
    size_t base = (((size_t)kb * NCHUNK) + c) * D_INNER + d;
    float local = ws_h[base * D_STATE + s];
    ws_h[base * D_STATE + s] = h;                      // carry-in for this chunk
    h = __expf(As * ws_dtsum[base]) * h + local;       // exp(A*sum_dt)*h_in + h_local
  }
}

// ---- pass C: recompute front-end, full scan with carry-in, gate, out-proj ----
__global__ void k_chunk_out(const float* x, const float* in_w, const float* conv_w,
                            const float* conv_b, const float* xproj_w,
                            const float* dt_w, const float* dt_b, const float* A_log,
                            const float* Dp, const float* outp_w,
                            const float* ws_h, __bf16* ws_Y) {
  extern __shared__ char smem[];
  float*  s_dt   = (float*)(smem + SM_DT);
  float*  s_xdbl = (float*)(smem + SM_XDBL);
  __bf16* s_xbf  = (__bf16*)(smem + SM_XBF);
  __bf16* s_xc   = (__bf16*)(smem + SM_XC);
  __bf16* s_u    = (__bf16*)(smem + SM_U);
  __bf16* s_z    = (__bf16*)(smem + SM_Z);

  int c = blockIdx.x, kb = blockIdx.y;
  int k = kb >> 1, b = kb & 1;
  chunk_frontend(k, b, c, true, x, in_w, conv_w, conv_b, xproj_w, dt_w, dt_b,
                 s_xbf, s_xc, s_u, s_dt, s_xdbl, s_z);

  int d = threadIdx.x;
  if (d < D_INNER) {
    float As[D_STATE], h[D_STATE];
    size_t base = (((size_t)kb * NCHUNK) + c) * D_INNER + d;
#pragma unroll
    for (int s = 0; s < D_STATE; ++s) {
      As[s] = -__expf(A_log[((size_t)k * D_INNER + d) * D_STATE + s]);
      h[s] = ws_h[base * D_STATE + s];
    }
    float Dpd = Dp[k * D_INNER + d];
    for (int t = 0; t < CHUNK; ++t) {
      float dtv = s_dt[t * D_INNER + d];
      float uv  = (float)s_u[t * D_INNER + d];
      float du  = dtv * uv;
      float y = 0.0f;
#pragma unroll
      for (int s = 0; s < D_STATE; ++s) {
        float Bs = s_xdbl[t * NDP + DT_RANK + s];
        float Cs = s_xdbl[t * NDP + DT_RANK + D_STATE + s];
        h[s] = __expf(dtv * As[s]) * h[s] + du * Bs;
        y += h[s] * Cs;
      }
      y += Dpd * uv;
      y *= (float)s_z[t * D_INNER + d];
      s_u[t * D_INNER + d] = (__bf16)y;  // reuse u buffer as gated y (safe: same slot)
    }
  }
  __syncthreads();

  // out-proj GEMM: (64x192 bf16) @ (192 -> 96), scatter to un-permuted Y buffer
  const float* Wo = outp_w + (size_t)k * D_MODEL * D_INNER;
  int lane = threadIdx.x & 31, wv = threadIdx.x >> 5;
  for (int j = wv; j < 24; j += 8) {
    int mt = j / 6, nt = j % 6;
    v8f acc = {};
#pragma unroll
    for (int kt = 0; kt < 6; ++kt) {
      v16bf a  = load_afrag_lds(s_u, D_INNER, mt * 16, kt * 32, lane);
      v16bf bb = load_bfrag_w(Wo, D_INNER, D_MODEL, nt * 16, kt * 32, lane);
      acc = wmma_bf16(a, bb, acc);
    }
    int M0 = (lane >> 4) << 3, N = lane & 15;
#pragma unroll
    for (int r = 0; r < 8; ++r) {
      int t = mt * 16 + M0 + r;
      int p = permP(k, c * CHUNK + t);
      ws_Y[((size_t)b * L + p) * 384 + k * 96 + nt * 16 + N] = (__bf16)acc[r];
    }
  }
}

// ---- pass D: fuse GEMM (B*L x 384) @ (384 -> 96) + bias, transposed store ----
__global__ void k_fuse(const __bf16* ws_Y, const float* fuse_w, const float* fuse_b,
                       float* out) {
  int lane = threadIdx.x & 31, wv = threadIdx.x >> 5;
  int mbase = blockIdx.x * 64;
  for (int j = wv; j < 24; j += 8) {
    int mt = j / 6, nt = j % 6;
    v8f acc = {};
#pragma unroll
    for (int kt = 0; kt < 12; ++kt) {
      const __bf16* p = ws_Y + ((size_t)(mbase + mt * 16 + (lane & 15))) * 384
                             + kt * 32 + ((lane >> 4) << 3);
      v16bf a;
#pragma unroll
      for (int e = 0; e < 8; ++e) { a[e] = p[e]; a[e + 8] = p[e + 16]; }
      v16bf bb = load_bfrag_w(fuse_w, 384, D_MODEL, nt * 16, kt * 32, lane);
      acc = wmma_bf16(a, bb, acc);
    }
    int M0 = (lane >> 4) << 3, N = lane & 15;
    int ch = nt * 16 + N;
    float fb = fuse_b[ch];
#pragma unroll
    for (int r = 0; r < 8; ++r) {
      int m = mbase + mt * 16 + M0 + r;
      int b = m >> 14, p = m & (L - 1);
      out[(((size_t)b * D_MODEL) + ch) * L + p] = acc[r] + fb;
    }
  }
}

extern "C" void kernel_launch(void* const* d_in, const int* in_sizes, int n_in,
                              void* d_out, int out_size, void* d_ws, size_t ws_size,
                              hipStream_t stream) {
  const float* x       = (const float*)d_in[0];
  const float* in_w    = (const float*)d_in[1];
  const float* conv_w  = (const float*)d_in[2];
  const float* conv_b  = (const float*)d_in[3];
  const float* xproj_w = (const float*)d_in[4];
  const float* dt_w    = (const float*)d_in[5];
  const float* dt_b    = (const float*)d_in[6];
  const float* A_log   = (const float*)d_in[7];
  const float* Dp      = (const float*)d_in[8];
  const float* outp_w  = (const float*)d_in[9];
  const float* fuse_w  = (const float*)d_in[10];
  const float* fuse_b  = (const float*)d_in[11];

  char* ws = (char*)d_ws;
  float*  ws_h     = (float*)(ws + WS_H_OFF);
  float*  ws_dtsum = (float*)(ws + WS_DT_OFF);
  __bf16* ws_Y     = (__bf16*)(ws + WS_Y_OFF);

  (void)hipFuncSetAttribute((const void*)k_chunk_state,
                            hipFuncAttributeMaxDynamicSharedMemorySize, SMEM_A);
  (void)hipFuncSetAttribute((const void*)k_chunk_out,
                            hipFuncAttributeMaxDynamicSharedMemorySize, SMEM_C);

  dim3 grid(NCHUNK, 4 * BATCH);
  k_chunk_state<<<grid, 256, SMEM_A, stream>>>(x, in_w, conv_w, conv_b, xproj_w,
                                               dt_w, dt_b, A_log, ws_h, ws_dtsum);
  k_combine<<<(8 * D_INNER * D_STATE + 255) / 256, 256, 0, stream>>>(A_log, ws_h, ws_dtsum);
  k_chunk_out<<<grid, 256, SMEM_C, stream>>>(x, in_w, conv_w, conv_b, xproj_w,
                                             dt_w, dt_b, A_log, Dp, outp_w, ws_h, ws_Y);
  k_fuse<<<(BATCH * L) / 64, 256, 0, stream>>>(ws_Y, fuse_w, fuse_b, (float*)d_out);
}
